// MultiHeadLinearAttention2D_51951924412913
// MI455X (gfx1250) — matrix operand order repaired and checked by reading, not compiled, not run on hardware
//
#include <hip/hip_runtime.h>
#include <hip/hip_bf16.h>
#include <math.h>

// Problem constants (from reference)
#define BATCH 16
#define HEADS 8
#define CIN   256
#define COUT  512
#define DH    64
#define MTOT  4096          // H*W = 64*64
#define TILE  128           // positions per tile
#define NTILE (MTOT / TILE) // 32

// LDS strides (bf16 elements), padded to dodge bank conflicts, 16B-aligned rows
#define SW_ST  264          // weight rows: 256 + 8 pad
#define SXT_ST 264          // x tile rows (position-major): 256 chan + 8 pad
#define SK_ST  136          // k/v tile rows (channel-major): 128 pos + 8 pad
#define SQT_ST 72           // q tile rows (position-major): 64 chan + 8 pad
#define KVT_ST 72           // KV^T rows: 64 + 8 pad

// LDS byte layout
#define SW_BYTES   (3 * DH * SW_ST * 2)    // 101376: Wq, Wk, Wv slices
#define SXT_BYTES  (TILE * SXT_ST * 2)     // 67584
#define SKV_BYTES  (2 * DH * SK_ST * 2)    // 34816 (k + v tiles; reused as sQT)
#define KVT_BYTES  (DH * KVT_ST * 2)       // 9216
#define KSUM_BYTES (DH * 4)                // 256
#define DEN_BYTES  (TILE * 4)              // 512
#define SMEM_TOTAL (SW_BYTES + SXT_BYTES + SKV_BYTES + KVT_BYTES + KSUM_BYTES + DEN_BYTES)
static_assert((TILE * SQT_ST * 2) <= SKV_BYTES, "sQT must fit in k/v region");
static_assert(SMEM_TOTAL <= 320 * 1024, "exceeds 320KB WGP LDS");

typedef __attribute__((ext_vector_type(16))) __bf16 v16bf;
typedef __attribute__((ext_vector_type(8)))  float  v8f;

// A fragment (16x32, M x K), ISA 7.12.2 16-bit A layout:
// lane l<16: M=l, V0..3 hold K=k0..k0+7, V4..7 hold K=k0+16..k0+23
// lane l>=16: same M, K offset +8 in each half.  Contiguous -> 2x ds_load_b128.
__device__ __forceinline__ v16bf frag_A(const __bf16* base, int stride,
                                        int row0, int k0, int lane) {
    const int l = lane & 15, hi = lane >> 4;
    const __bf16* rp = base + (row0 + l) * stride + k0 + hi * 8;
    v16bf a;
#pragma unroll
    for (int e = 0; e < 8; ++e) a[e] = rp[e];
#pragma unroll
    for (int e = 0; e < 8; ++e) a[8 + e] = rp[16 + e];
    return a;
}

// B fragment (32x16, K x N) where B[k][n] is stored transposed in LDS
// (mem[n][k] row-major): lane owns column n = n0+(lane&15); same K split as
// frag_A: elems 0..7 K=k0+8*hi.., elems 8..15 K=k0+16+8*hi.. -> 2x ds_load_b128.
__device__ __forceinline__ v16bf frag_B_row(const __bf16* base, int stride,
                                            int n0, int k0, int lane) {
    const int l = lane & 15, hi = lane >> 4;
    const __bf16* p = base + (n0 + l) * stride + k0 + hi * 8;
    v16bf b;
#pragma unroll
    for (int e = 0; e < 8; ++e) b[e] = p[e];
#pragma unroll
    for (int e = 0; e < 8; ++e) b[8 + e] = p[16 + e];
    return b;
}

__device__ __forceinline__ v8f wmma_bf16(v16bf a, v16bf b, v8f c) {
    return __builtin_amdgcn_wmma_f32_16x16x32_bf16(false, a, false, b,
                                                   (short)0, c, false, false);
}

// Position-major x tile load. Thread owns a 4-position x 32-channel block:
// float4 loads along positions (lane-consecutive quads -> coalesced b128),
// packed 16-byte LDS stores per position row.
__device__ __forceinline__ void load_x_tile_T(const float* __restrict__ x, __bf16* sXT,
                                              int b, int m0, int t, bool prefetch_next) {
    const int qp  = t & 31;    // position quad: positions qp*4 .. qp*4+3
    const int oct = t >> 5;    // channel octant: channels oct*32 .. +31
    const float* xp = x + ((size_t)b * CIN + (size_t)oct * 32) * MTOT + m0 + qp * 4;
    __bf16* dst = sXT + (qp * 4) * SXT_ST + oct * 32;
#pragma unroll
    for (int cc = 0; cc < 32; cc += 8) {
        __bf16 buf[4][8];
#pragma unroll
        for (int c = 0; c < 8; ++c) {
            float4 v4 = *(const float4*)(xp + (size_t)(cc + c) * MTOT);
            buf[0][c] = (__bf16)v4.x;
            buf[1][c] = (__bf16)v4.y;
            buf[2][c] = (__bf16)v4.z;
            buf[3][c] = (__bf16)v4.w;
        }
#pragma unroll
        for (int r = 0; r < 4; ++r) {
            *(uint4*)(dst + r * SXT_ST + cc) = *(const uint4*)buf[r];
        }
    }
    if (prefetch_next) __builtin_prefetch(xp + TILE, 0, 0);  // global_prefetch_b8
}

__global__ void __launch_bounds__(256, 1)
mhla2d_fused_kernel(const float* __restrict__ x,  const float* __restrict__ Wq,
                    const float* __restrict__ Wk, const float* __restrict__ Wv,
                    float* __restrict__ out) {
    extern __shared__ char smem[];
    __bf16* sW    = (__bf16*)(smem);                         // [3][DH][SW_ST] q,k,v
    __bf16* sXT   = (__bf16*)(smem + SW_BYTES);              // [TILE][SXT_ST] pos-major
    __bf16* sK    = (__bf16*)(smem + SW_BYTES + SXT_BYTES);  // [DH][SK_ST] chan-major
    __bf16* sV    = sK + DH * SK_ST;                         // [DH][SK_ST]
    __bf16* sQT   = sK;                                      // [TILE][SQT_ST] (phase 2)
    __bf16* sKVT  = (__bf16*)(smem + SW_BYTES + SXT_BYTES + SKV_BYTES);
    float*  sKsum = (float*)(smem + SW_BYTES + SXT_BYTES + SKV_BYTES + KVT_BYTES);
    float*  sDen  = (float*)(smem + SW_BYTES + SXT_BYTES + SKV_BYTES + KVT_BYTES + KSUM_BYTES);

    const int b = blockIdx.x >> 3;
    const int h = blockIdx.x & 7;
    const int t = threadIdx.x;
    const int lane = t & 31;          // wave32
    const int w = t >> 5;             // wave id 0..7 -> 16-wide position strip
    const int l = lane & 15, hi = lane >> 4;
    const float scale = 0.125f;       // 1/sqrt(dh=64)
    const float LOG1E4 = 9.2103403719761836f;

    // ---- stage head's weight slices into LDS as bf16 (packed 8B stores) ----
    {
        const int row = t >> 2, seg = (t & 3) * 64;
        const size_t off = (size_t)(h * DH + row) * CIN + seg;
        const float* sq = Wq + off;
        const float* sk = Wk + off;
        const float* sv = Wv + off;
        __bf16* dq = sW + 0 * DH * SW_ST + row * SW_ST + seg;
        __bf16* dk = sW + 1 * DH * SW_ST + row * SW_ST + seg;
        __bf16* dv = sW + 2 * DH * SW_ST + row * SW_ST + seg;
#pragma unroll
        for (int i = 0; i < 64; i += 4) {
            float4 a = *(const float4*)(sq + i);
            __bf16 ha[4] = {(__bf16)a.x, (__bf16)a.y, (__bf16)a.z, (__bf16)a.w};
            *(uint2*)(dq + i) = *(const uint2*)ha;
            float4 c = *(const float4*)(sk + i);
            __bf16 hc[4] = {(__bf16)c.x, (__bf16)c.y, (__bf16)c.z, (__bf16)c.w};
            *(uint2*)(dk + i) = *(const uint2*)hc;
            float4 d = *(const float4*)(sv + i);
            __bf16 hd[4] = {(__bf16)d.x, (__bf16)d.y, (__bf16)d.z, (__bf16)d.w};
            *(uint2*)(dv + i) = *(const uint2*)hd;
        }
    }
    if (t < DH) sKsum[t] = 0.0f;
    __syncthreads();

    // ================= Phase 1: KV = phi(k) v^T, k_sum =================
    v8f kv0 = {};
    v8f kv1 = {};
    v8f ksacc[4];       // register accumulation of k_sum partials
    {
        v8f zero = {};
#pragma unroll
        for (int mt = 0; mt < 4; ++mt) ksacc[mt] = zero;
    }
    const int ct0 = (2 * w) >> 2,     dt0 = (2 * w) & 3;
    const int ct1 = (2 * w + 1) >> 2, dt1 = (2 * w + 1) & 3;

    for (int tIdx = 0; tIdx < NTILE; ++tIdx) {
        const int m0 = tIdx * TILE;
        load_x_tile_T(x, sXT, b, m0, t, tIdx + 1 < NTILE);
        __syncthreads();

        // k = Wk_h @ x ; v = Wv_h @ x  (each wave: 16-wide strip, 4 row tiles)
        v8f ak[4], av[4];
        v8f zero = {};
#pragma unroll
        for (int mt = 0; mt < 4; ++mt) { ak[mt] = zero; av[mt] = zero; }
#pragma unroll
        for (int ks = 0; ks < 8; ++ks) {
            v16bf bx = frag_B_row(sXT, SXT_ST, w * 16, ks * 32, lane);
#pragma unroll
            for (int mt = 0; mt < 4; ++mt) {
                v16bf aK = frag_A(sW + 1 * DH * SW_ST, SW_ST, mt * 16, ks * 32, lane);
                ak[mt] = wmma_bf16(aK, bx, ak[mt]);
                v16bf aV = frag_A(sW + 2 * DH * SW_ST, SW_ST, mt * 16, ks * 32, lane);
                av[mt] = wmma_bf16(aV, bx, av[mt]);
            }
        }
        // phi_k = (elu(k)+1)*scale = (k>0 ? k+1 : exp(k))*scale ; write tiles
#pragma unroll
        for (int mt = 0; mt < 4; ++mt) {
#pragma unroll
            for (int e = 0; e < 8; ++e) {
                const int row = mt * 16 + hi * 8 + e;
                const int col = w * 16 + l;
                float kf = ak[mt][e];
                kf = (kf > 0.0f ? kf + 1.0f : __expf(kf)) * scale;
                sK[row * SK_ST + col] = (__bf16)kf;
                ksacc[mt][e] += kf;
                sV[row * SK_ST + col] = (__bf16)av[mt][e];
            }
        }
        __syncthreads();

        // KV += k_tile(64xT) @ v_tile^T(Tx64); 2 of 16 output tiles per wave
#pragma unroll
        for (int ks = 0; ks < 4; ++ks) {
            v16bf aK0 = frag_A(sK, SK_ST, ct0 * 16, ks * 32, lane);
            v16bf bV0 = frag_B_row(sV, SK_ST, dt0 * 16, ks * 32, lane);
            kv0 = wmma_bf16(aK0, bV0, kv0);
            v16bf aK1 = frag_A(sK, SK_ST, ct1 * 16, ks * 32, lane);
            v16bf bV1 = frag_B_row(sV, SK_ST, dt1 * 16, ks * 32, lane);
            kv1 = wmma_bf16(aK1, bV1, kv1);
        }
        __syncthreads();
    }

    // flush k_sum register partials (once, not per tile)
#pragma unroll
    for (int mt = 0; mt < 4; ++mt) {
#pragma unroll
        for (int e = 0; e < 8; ++e) {
            atomicAdd(&sKsum[mt * 16 + hi * 8 + e], ksacc[mt][e]);
        }
    }

    // KV^T into LDS as bf16 (rows d, cols c) for phase-2 A operand
#pragma unroll
    for (int i = 0; i < 2; ++i) {
        const int ct = i ? ct1 : ct0;
        const int dt = i ? dt1 : dt0;
        v8f acc = i ? kv1 : kv0;
#pragma unroll
        for (int e = 0; e < 8; ++e) {
            const int c = ct * 16 + hi * 8 + e;
            const int d = dt * 16 + l;
            sKVT[d * KVT_ST + c] = (__bf16)acc[e];
        }
    }
    __syncthreads();

    // ================= Phase 2: q -> out = KV^T q / (q . k_sum) =================
    for (int tIdx = 0; tIdx < NTILE; ++tIdx) {
        const int m0 = tIdx * TILE;
        load_x_tile_T(x, sXT, b, m0, t, tIdx + 1 < NTILE);
        __syncthreads();

        // q = Wq_h @ x
        v8f aq[4];
        v8f zero = {};
#pragma unroll
        for (int mt = 0; mt < 4; ++mt) aq[mt] = zero;
#pragma unroll
        for (int ks = 0; ks < 8; ++ks) {
            v16bf bx = frag_B_row(sXT, SXT_ST, w * 16, ks * 32, lane);
#pragma unroll
            for (int mt = 0; mt < 4; ++mt) {
                v16bf aQ = frag_A(sW, SW_ST, mt * 16, ks * 32, lane);
                aq[mt] = wmma_bf16(aQ, bx, aq[mt]);
            }
        }
        // + 2D sinusoidal PE, phi_q; store q tile position-major.
        // Each lane's 8 e-values are 8 consecutive channels of one position
        // -> single 16-byte LDS store per (mt).
        const int col = w * 16 + l;
        const int m = m0 + col;
        const int pi = m >> 6, pj = m & 63;           // H = W = 64
        const float pos = (h < 4) ? (float)pi : (float)pj;
#pragma unroll
        for (int mt = 0; mt < 4; ++mt) {
            __bf16 qb[8];
#pragma unroll
            for (int e = 0; e < 8; ++e) {
                const int row = mt * 16 + hi * 8 + e;     // channel within head
                const int c = (h & 3) * 64 + row;         // channel within half (256)
                const float freq = __expf(-LOG1E4 * (float)(c & ~1) * (1.0f / 256.0f));
                const float ang = pos * freq;
                const float r = (c & 1) ? __cosf(ang) : __sinf(ang);
                float qf = aq[mt][e] + r;
                qf = (qf > 0.0f ? qf + 1.0f : __expf(qf)) * scale;
                qb[e] = (__bf16)qf;
            }
            *(uint4*)(sQT + col * SQT_ST + mt * 16 + hi * 8) = *(const uint4*)qb;
        }
        __syncthreads();

        // denominators per position (contiguous per-thread read of sQT row)
        if (t < TILE) {
            const __bf16* qr = sQT + t * SQT_ST;
            float s = 0.0f;
#pragma unroll
            for (int c = 0; c < DH; ++c) s += (float)qr[c] * sKsum[c];
            sDen[t] = fmaxf(s, 1e-6f);
        }
        __syncthreads();

        // out_tile = KV^T(64x64) @ q_tile(64xT)
        v8f ao[4];
#pragma unroll
        for (int dt = 0; dt < 4; ++dt) ao[dt] = zero;
#pragma unroll
        for (int ks = 0; ks < 2; ++ks) {
            v16bf bq = frag_B_row(sQT, SQT_ST, w * 16, ks * 32, lane);
#pragma unroll
            for (int dt = 0; dt < 4; ++dt) {
                v16bf aKV = frag_A(sKVT, KVT_ST, dt * 16, ks * 32, lane);
                ao[dt] = wmma_bf16(aKV, bq, ao[dt]);
            }
        }
        const float invd = 1.0f / sDen[col];
        float* op = out + ((size_t)b * COUT + (size_t)h * DH) * MTOT + m0 + col;
#pragma unroll
        for (int dt = 0; dt < 4; ++dt) {
#pragma unroll
            for (int e = 0; e < 8; ++e) {
                const int d = dt * 16 + hi * 8 + e;
                op[(size_t)d * MTOT] = ao[dt][e] * invd;
            }
        }
        __syncthreads();
    }
}

extern "C" void kernel_launch(void* const* d_in, const int* in_sizes, int n_in,
                              void* d_out, int out_size, void* d_ws, size_t ws_size,
                              hipStream_t stream) {
    (void)in_sizes; (void)n_in; (void)out_size; (void)d_ws; (void)ws_size;
    const float* x  = (const float*)d_in[0];
    const float* Wq = (const float*)d_in[1];
    const float* Wk = (const float*)d_in[2];
    const float* Wv = (const float*)d_in[3];
    float* out = (float*)d_out;

    // Allow >64KB dynamic LDS (CDNA5 WGP has 320KB); deterministic, capture-safe.
    (void)hipFuncSetAttribute((const void*)mhla2d_fused_kernel,
                              hipFuncAttributeMaxDynamicSharedMemorySize, SMEM_TOTAL);

    dim3 grid(BATCH * HEADS);   // 128 workgroups, one (b,h) pair each
    dim3 block(256);            // 8 wave32
    mhla2d_fused_kernel<<<grid, block, SMEM_TOTAL, stream>>>(x, Wq, Wk, Wv, out);
}
